// BalancedMoEFFN_11441792877182
// MI455X (gfx1250) — matrix operand order repaired and checked by reading, not compiled, not run on hardware
//
#include <hip/hip_runtime.h>
#include <hip/hip_bf16.h>
#include <math.h>

#define D_MODEL 1024
#define D_FF    2048
#define N_EXP   8
#define NTOK    8192                 // B*S = 4*2048
#define NROWS   (NTOK * 2)           // token-expert pairs (top-2)

typedef __attribute__((ext_vector_type(16))) __bf16 bf16x16;
typedef __attribute__((ext_vector_type(8)))  __bf16 bf16x8;
typedef __attribute__((ext_vector_type(4)))  __bf16 bf16x4;
typedef __attribute__((ext_vector_type(8)))  float  f32x8;
typedef __attribute__((ext_vector_type(4)))  int    v4i;
typedef __attribute__((address_space(1)))    v4i    gv4i;   // global int4
typedef __attribute__((address_space(3)))    v4i    lv4i;   // LDS int4

union AFrag { bf16x16 v; struct { bf16x8 lo; bf16x8 hi; } p; };

// ---- CDNA5 async global->LDS staging (ASYNCcnt path), guarded fallback ----
#if __has_builtin(__builtin_amdgcn_global_load_async_to_lds_b128)
#define ASYNC_CP16(dst, src)                                                   \
    __builtin_amdgcn_global_load_async_to_lds_b128((gv4i*)(src), (lv4i*)(dst), \
                                                   0, 0)
#define ASYNC_WAIT() asm volatile("s_wait_asynccnt 0x0" ::: "memory")
#else
#define ASYNC_CP16(dst, src) (*(bf16x8*)(dst) = *(const bf16x8*)(src))
#define ASYNC_WAIT()
#endif

// ---------------- init: zero output + meta counters ----------------
__global__ void moe_init(float* out, int n, int* meta) {
    for (int i = blockIdx.x * blockDim.x + threadIdx.x; i < n;
         i += gridDim.x * blockDim.x)
        out[i] = 0.0f;
    if (blockIdx.x == 0 && threadIdx.x < 32) meta[threadIdx.x] = 0;
}

// ------- one-time weight convert+transpose: (R,C) f32 -> (C,R) bf16 -------
__global__ __launch_bounds__(256)
void moe_cvtw(const float* __restrict__ src, __bf16* __restrict__ dst,
              int R, int C) {
    const int e = blockIdx.z;
    const float* s = src + (size_t)e * R * C;
    __bf16* d = dst + (size_t)e * R * C;
    const int cb = blockIdx.x * 32;   // C-dim tile base
    const int rb = blockIdx.y * 32;   // R-dim tile base
    __shared__ __bf16 T[32][40];      // [c][r], padded row (80B, 8B-aligned)
    const int t = threadIdx.x;
    {
        const int r = t >> 3, c4 = (t & 7) * 4;
        const float4 v = *(const float4*)(s + (size_t)(rb + r) * C + cb + c4);
        T[c4 + 0][r] = (__bf16)v.x;  T[c4 + 1][r] = (__bf16)v.y;
        T[c4 + 2][r] = (__bf16)v.z;  T[c4 + 3][r] = (__bf16)v.w;
    }
    __syncthreads();
    {
        const int c = t >> 3, r4 = (t & 7) * 4;
        *(bf16x4*)(d + (size_t)(cb + c) * R + rb + r4) = *(const bf16x4*)&T[c][r4];
    }
}

// ---------------- router: logits, top-2, x -> bf16 ----------------
__global__ void moe_router(const float* __restrict__ x,
                           const float* __restrict__ rw,
                           const float* __restrict__ bias,
                           __bf16* __restrict__ xb,
                           int* __restrict__ topk_i,
                           float* __restrict__ topk_w,
                           int* __restrict__ meta) {
    const int wave = threadIdx.x >> 5;
    const int lane = threadIdx.x & 31;
    const int token = blockIdx.x * 8 + wave;
    if (token >= NTOK) return;
    const float* xr = x + (size_t)token * D_MODEL;
    float s[N_EXP];
#pragma unroll
    for (int e = 0; e < N_EXP; ++e) s[e] = 0.0f;
    for (int i = 0; i < D_MODEL / 32; ++i) {
        const int d = lane + i * 32;
        const float xv = xr[d];
        xb[(size_t)token * D_MODEL + d] = (__bf16)xv;
#pragma unroll
        for (int e = 0; e < N_EXP; ++e) s[e] += xv * rw[e * D_MODEL + d];
    }
#pragma unroll
    for (int e = 0; e < N_EXP; ++e)
        for (int m = 16; m; m >>= 1) s[e] += __shfl_xor(s[e], m, 32);
    if (lane == 0) {
        float l[N_EXP];
#pragma unroll
        for (int e = 0; e < N_EXP; ++e) l[e] = s[e] + bias[e];
        int i1 = 0;
#pragma unroll
        for (int e = 1; e < N_EXP; ++e) if (l[e] > l[i1]) i1 = e;
        int i2 = (i1 == 0) ? 1 : 0;
#pragma unroll
        for (int e = 0; e < N_EXP; ++e)
            if (e != i1 && l[e] > l[i2]) i2 = e;
        // softmax denominator cancels under top-k renormalization
        const float p2 = __expf(l[i2] - l[i1]);
        const float inv = 1.0f / (1.0f + p2);
        topk_i[token * 2 + 0] = i1;  topk_w[token * 2 + 0] = inv;
        topk_i[token * 2 + 1] = i2;  topk_w[token * 2 + 1] = p2 * inv;
        atomicAdd(&meta[i1], 1);
        atomicAdd(&meta[i2], 1);
    }
}

// ---------------- scan: counts -> offsets / cursors ----------------
__global__ void moe_scan(int* meta) {
    if (blockIdx.x == 0 && threadIdx.x == 0) {
        int acc = 0;
        for (int e = 0; e < N_EXP; ++e) {
            meta[8 + e]  = acc;   // offset
            meta[16 + e] = acc;   // cursor
            acc += meta[e];
        }
    }
}

// ---------------- fill: scatter tokens into per-expert lists ----------------
__global__ void moe_fill(const int* __restrict__ topk_i,
                         const float* __restrict__ topk_w,
                         int* __restrict__ meta,
                         int* __restrict__ rows,
                         float* __restrict__ rwt) {
    const int t = blockIdx.x * blockDim.x + threadIdx.x;
    if (t >= NTOK) return;
#pragma unroll
    for (int k = 0; k < 2; ++k) {
        const int e = topk_i[t * 2 + k];
        const int pos = atomicAdd(&meta[16 + e], 1);
        rows[pos] = t;
        rwt[pos] = topk_w[t * 2 + k];
    }
}

// ---------------- GEMM1: H = silu(X Wg) * (X Wu), bf16 out ----------------
// wgT/wuT are (E, F, D) bf16: B tiles are row-major [n][k] -> direct async copy
__global__ __launch_bounds__(256, 2)
void moe_gemm1(const __bf16* __restrict__ xb,
               const __bf16* __restrict__ wgT,
               const __bf16* __restrict__ wuT,
               const int* __restrict__ rows,
               const int* __restrict__ meta,
               __bf16* __restrict__ H) {
    const int e     = blockIdx.z;
    const int cnt   = meta[e];
    const int mbase = blockIdx.y * 64;
    if (mbase >= cnt) return;
    const int off   = meta[8 + e];
    const int fbase = blockIdx.x * 128;

    __shared__ __bf16 Xs[64][64];     // [row][k]
    __shared__ __bf16 Gs[128][64];    // [f][k]
    __shared__ __bf16 Us[128][64];
    __shared__ int ridx[64];

    const int tid = threadIdx.x;
    if (tid < 64) {
        const int r = mbase + tid;
        ridx[tid] = rows[off + ((r < cnt) ? r : 0)];
    }
    __syncthreads();

    const size_t wbase = (size_t)e * D_MODEL * D_FF;
    f32x8 accg[4] = {};
    f32x8 accu[4] = {};
    const int lane = tid & 31, wave = tid >> 5;
    const int wm = (wave & 3) * 16;      // wave M block
    const int wn = (wave >> 2) * 64;     // wave N block
    const int lm = lane & 15, lh = lane >> 4;

    for (int kk = 0; kk < D_MODEL; kk += 64) {
        // stage gathered X tile (64 x 64 bf16) via async LDS copies
#pragma unroll
        for (int i = 0; i < 2; ++i) {
            const int l = tid + i * 256, row = l >> 3, seg = l & 7;
            ASYNC_CP16(&Xs[row][seg * 8],
                       xb + (size_t)ridx[row] * D_MODEL + kk + seg * 8);
        }
        // stage gate/up weight tiles (128 x 64 bf16, already transposed)
#pragma unroll
        for (int i = 0; i < 4; ++i) {
            const int l = tid + i * 256, row = l >> 3, seg = l & 7;
            const __bf16* gs = wgT + wbase + (size_t)(fbase + row) * D_MODEL + kk + seg * 8;
            const __bf16* us = wuT + wbase + (size_t)(fbase + row) * D_MODEL + kk + seg * 8;
            ASYNC_CP16(&Gs[row][seg * 8], gs);
            ASYNC_CP16(&Us[row][seg * 8], us);
            if (kk + 64 < D_MODEL) __builtin_prefetch(gs + 64, 0, 1);
        }
        ASYNC_WAIT();
        __syncthreads();

#pragma unroll
        for (int ks = 0; ks < 2; ++ks) {
            // batch all fragment loads first, then issue WMMAs back-to-back
            AFrag a;
            a.p.lo = *(const bf16x8*)&Xs[wm + lm][ks * 32 + lh * 16];
            a.p.hi = *(const bf16x8*)&Xs[wm + lm][ks * 32 + lh * 16 + 8];
            bf16x16 bg[4], bu[4];
#pragma unroll
            for (int j = 0; j < 4; ++j) {
                bg[j] = *(const bf16x16*)&Gs[wn + j * 16 + lm][ks * 32 + lh * 16];
                bu[j] = *(const bf16x16*)&Us[wn + j * 16 + lm][ks * 32 + lh * 16];
            }
#pragma unroll
            for (int j = 0; j < 4; ++j) {
                accg[j] = __builtin_amdgcn_wmma_f32_16x16x32_bf16(
                    false, a.v, false, bg[j], (short)0, accg[j], false, false);
                accu[j] = __builtin_amdgcn_wmma_f32_16x16x32_bf16(
                    false, a.v, false, bu[j], (short)0, accu[j], false, false);
            }
        }
        __syncthreads();
    }

    // epilogue: silu(gate) * up -> H (bf16)
#pragma unroll
    for (int j = 0; j < 4; ++j)
#pragma unroll
        for (int v = 0; v < 8; ++v) {
            const int mrow = v + 8 * lh;
            const int grow = mbase + wm + mrow;
            if (grow < cnt) {
                const float g = accg[j][v];
                const float u = accu[j][v];
                const float hv = (g / (1.0f + __expf(-g))) * u;
                H[(size_t)(off + grow) * D_FF + fbase + wn + j * 16 + lm] = (__bf16)hv;
            }
        }
}

// ---------------- GEMM2: out += rw * (H Wd) ----------------
// wdT is (E, D, F) bf16: B tiles row-major [n][k] over k=F
__global__ __launch_bounds__(256, 2)
void moe_gemm2(const __bf16* __restrict__ H,
               const __bf16* __restrict__ wdT,
               const int* __restrict__ rows,
               const float* __restrict__ rwt,
               const int* __restrict__ meta,
               float* __restrict__ out) {
    const int e     = blockIdx.z;
    const int cnt   = meta[e];
    const int mbase = blockIdx.y * 64;
    if (mbase >= cnt) return;
    const int off   = meta[8 + e];
    const int nbase = blockIdx.x * 128;

    __shared__ __bf16 As[64][64];     // [row][k]
    __shared__ __bf16 Bs[128][64];    // [d][k]

    const int tid = threadIdx.x;
    const size_t wbase = (size_t)e * D_FF * D_MODEL;
    f32x8 acc[4] = {};
    const int lane = tid & 31, wave = tid >> 5;
    const int wm = (wave & 3) * 16;
    const int wn = (wave >> 2) * 64;
    const int lm = lane & 15, lh = lane >> 4;

    for (int kk = 0; kk < D_FF; kk += 64) {
#pragma unroll
        for (int i = 0; i < 2; ++i) {
            const int l = tid + i * 256, row = l >> 3, seg = l & 7;
            ASYNC_CP16(&As[row][seg * 8],
                       H + (size_t)(off + mbase + row) * D_FF + kk + seg * 8);
        }
#pragma unroll
        for (int i = 0; i < 4; ++i) {
            const int l = tid + i * 256, row = l >> 3, seg = l & 7;
            const __bf16* bs = wdT + wbase + (size_t)(nbase + row) * D_FF + kk + seg * 8;
            ASYNC_CP16(&Bs[row][seg * 8], bs);
            if (kk + 64 < D_FF) __builtin_prefetch(bs + 64, 0, 1);
        }
        ASYNC_WAIT();
        __syncthreads();

#pragma unroll
        for (int ks = 0; ks < 2; ++ks) {
            AFrag a;
            a.p.lo = *(const bf16x8*)&As[wm + lm][ks * 32 + lh * 16];
            a.p.hi = *(const bf16x8*)&As[wm + lm][ks * 32 + lh * 16 + 8];
            bf16x16 b[4];
#pragma unroll
            for (int j = 0; j < 4; ++j)
                b[j] = *(const bf16x16*)&Bs[wn + j * 16 + lm][ks * 32 + lh * 16];
#pragma unroll
            for (int j = 0; j < 4; ++j)
                acc[j] = __builtin_amdgcn_wmma_f32_16x16x32_bf16(
                    false, a.v, false, b[j], (short)0, acc[j], false, false);
        }
        __syncthreads();
    }

#pragma unroll
    for (int j = 0; j < 4; ++j)
#pragma unroll
        for (int v = 0; v < 8; ++v) {
            const int mrow = v + 8 * lh;
            const int grow = mbase + wm + mrow;
            if (grow < cnt) {
                const float w = rwt[off + grow];
                const int tok = rows[off + grow];
                const int col = nbase + wn + j * 16 + lm;
                atomicAdd(&out[(size_t)tok * D_MODEL + col], w * acc[j][v]);
            }
        }
}

// ---------------- host launcher ----------------
extern "C" void kernel_launch(void* const* d_in, const int* in_sizes, int n_in,
                              void* d_out, int out_size, void* d_ws, size_t ws_size,
                              hipStream_t stream) {
    (void)in_sizes; (void)n_in; (void)ws_size;
    const float* x    = (const float*)d_in[0];
    const float* rw   = (const float*)d_in[1];
    const float* bias = (const float*)d_in[2];
    const float* wg   = (const float*)d_in[3];
    const float* wu   = (const float*)d_in[4];
    const float* wd   = (const float*)d_in[5];
    float* out = (float*)d_out;

    // workspace carve-up (all chunks 256B-aligned by construction)
    char* w = (char*)d_ws;
    __bf16* xb  = (__bf16*)w;           w += (size_t)NTOK * D_MODEL * 2;        // 16 MB
    __bf16* H   = (__bf16*)w;           w += ((size_t)NROWS + 64) * D_FF * 2;   // ~67 MB (+pad)
    __bf16* wgT = (__bf16*)w;           w += (size_t)N_EXP * D_MODEL * D_FF * 2; // 33.5 MB
    __bf16* wuT = (__bf16*)w;           w += (size_t)N_EXP * D_MODEL * D_FF * 2; // 33.5 MB
    __bf16* wdT = (__bf16*)w;           w += (size_t)N_EXP * D_MODEL * D_FF * 2; // 33.5 MB
    int*   topk_i = (int*)w;            w += (size_t)NTOK * 2 * 4;
    float* topk_w = (float*)w;          w += (size_t)NTOK * 2 * 4;
    int*   rows   = (int*)w;            w += (size_t)NROWS * 4;
    float* rwt    = (float*)w;          w += (size_t)NROWS * 4;
    int*   meta   = (int*)w;            w += 256;  // cnt[8], off[8], cur[8]

    moe_init<<<2048, 256, 0, stream>>>(out, out_size, meta);
    // wg (E,D,F) -> wgT (E,F,D); wu likewise; wd (E,F,D) -> wdT (E,D,F)
    moe_cvtw<<<dim3(D_FF / 32, D_MODEL / 32, N_EXP), 256, 0, stream>>>(wg, wgT, D_MODEL, D_FF);
    moe_cvtw<<<dim3(D_FF / 32, D_MODEL / 32, N_EXP), 256, 0, stream>>>(wu, wuT, D_MODEL, D_FF);
    moe_cvtw<<<dim3(D_MODEL / 32, D_FF / 32, N_EXP), 256, 0, stream>>>(wd, wdT, D_FF, D_MODEL);
    moe_router<<<NTOK / 8, 256, 0, stream>>>(x, rw, bias, xb, topk_i, topk_w, meta);
    moe_scan<<<1, 32, 0, stream>>>(meta);
    moe_fill<<<NTOK / 256, 256, 0, stream>>>(topk_i, topk_w, meta, rows, rwt);
    moe_gemm1<<<dim3(D_FF / 128, 64, N_EXP), 256, 0, stream>>>(xb, wgT, wuT, rows, meta, H);
    moe_gemm2<<<dim3(D_MODEL / 128, 64, N_EXP), 256, 0, stream>>>(H, wdT, rows, rwt, meta, out);
}
